// StackLSTMDecoder_12807592476791
// MI455X (gfx1250) — matrix operand order repaired and checked by reading, not compiled, not run on hardware
//
#include <hip/hip_runtime.h>

#define H_DIM   1024
#define B_DIM   64
#define V_DIM   32000
#define NSTEPS  50
#define KCAT    2048          // concatenated [x | h] inner dim
#define G4      4096          // 4*H gate rows
#define LCOLS   128           // logits columns per workgroup
#define NPART   (V_DIM / LCOLS)  // 250 partial argmax blocks

typedef __attribute__((ext_vector_type(16))) __bf16 v16bf;
typedef __attribute__((ext_vector_type(8)))  float  v8f;
typedef unsigned short bf16_t;

__device__ __forceinline__ bf16_t f2bf(float f) {
  unsigned u = __float_as_uint(f);
  u += 0x7FFFu + ((u >> 16) & 1u);     // round-to-nearest-even
  return (bf16_t)(u >> 16);
}

union Frag16 { uint4 q[2]; v16bf v; };

// A-matrix 16x32 bf16, contiguous-K order shared with B: lane L holds row m0+(L&15),
// 16 contiguous K starting at (L>>4)*16.
__device__ __forceinline__ v16bf load_a_frag(const bf16_t* base, int ld,
                                             int m0, int k0, int lane) {
  const int row = m0 + (lane & 15);
  const int kh  = (lane >> 4) << 4;
  const bf16_t* p = base + (size_t)row * ld + k0 + kh;
  Frag16 f;
  f.q[0] = *(const uint4*)(p);
  f.q[1] = *(const uint4*)(p + 8);
  return f.v;
}

// B-matrix 32x16 bf16 (K x N), B[k][n] = W[n0+n][k0+k] with W row-major [N,K]:
// lane L holds column n0+(L&15), 16 contiguous K starting at k0+(L>>4)*16.
__device__ __forceinline__ v16bf load_b_frag(const bf16_t* w, int ld,
                                             int n0, int k0, int lane) {
  const int col = n0 + (lane & 15);
  const int ko  = k0 + ((lane >> 4) << 4);
  const bf16_t* p = w + (size_t)col * ld + ko;
  Frag16 f;
  f.q[0] = *(const uint4*)(p);
  f.q[1] = *(const uint4*)(p + 8);
  return f.v;
}

__device__ __forceinline__ v8f wmma_bf16(v16bf a, v16bf b, v8f c) {
  return __builtin_amdgcn_wmma_f32_16x16x32_bf16(
      /*neg_a=*/false, a, /*neg_b=*/false, b,
      /*c_mod=*/(short)0, c, /*reuse_a=*/false, /*reuse_b=*/false);
}

__device__ __forceinline__ float sigf(float x) { return 1.0f / (1.0f + __expf(-x)); }

// ---------------- weight / bias prep (amortized across 50 steps) -------------

__global__ void k_conv_wcat(const float* __restrict__ wih, const float* __restrict__ whh,
                            bf16_t* __restrict__ wcat) {
  const int n = blockIdx.x;                       // 0..4095 gate rows
  const float* s0 = wih + (size_t)n * H_DIM;
  const float* s1 = whh + (size_t)n * H_DIM;
  bf16_t* dst = wcat + (size_t)n * KCAT;
  for (int k = threadIdx.x; k < H_DIM; k += blockDim.x) {
    dst[k]         = f2bf(s0[k]);
    dst[H_DIM + k] = f2bf(s1[k]);
  }
}

__global__ void k_conv_wout(const float* __restrict__ w, bf16_t* __restrict__ o, int n) {
  for (size_t i = (size_t)blockIdx.x * blockDim.x + threadIdx.x; i < (size_t)n;
       i += (size_t)gridDim.x * blockDim.x)
    o[i] = f2bf(w[i]);
}

__global__ void k_bias(const float* a0, const float* a1, float* s0,
                       const float* c0, const float* c1, float* s1) {
  int n = blockIdx.x * blockDim.x + threadIdx.x;
  if (n < G4) { s0[n] = a0[n] + a1[n]; s1[n] = c0[n] + c1[n]; }
}

__global__ void k_init(const float* __restrict__ hidden, bf16_t* __restrict__ xh0,
                       bf16_t* __restrict__ xh1, float* __restrict__ c0,
                       float* __restrict__ c1, int* __restrict__ last) {
  int i = blockIdx.x * blockDim.x + threadIdx.x;   // covers B*H
  if (i < B_DIM * H_DIM) {
    int b = i >> 10, k = i & (H_DIM - 1);
    xh0[(size_t)b * KCAT + H_DIM + k] = f2bf(hidden[i]);                   // h0 prev
    xh1[(size_t)b * KCAT + H_DIM + k] = f2bf(hidden[B_DIM * H_DIM + i]);   // h1 prev
    c0[i] = 0.f; c1[i] = 0.f;
  }
  if (i < B_DIM) last[i] = 1;  // START_INDEX
}

// ---------------- per-step kernels -------------------------------------------

__global__ void k_gather(const float* __restrict__ emb, const int* __restrict__ last,
                         bf16_t* __restrict__ xh) {
  const int b = blockIdx.x;
  const float* src = emb + (size_t)last[b] * H_DIM;
  bf16_t* dst = xh + (size_t)b * KCAT;
  for (int j = threadIdx.x; j < H_DIM; j += blockDim.x) dst[j] = f2bf(src[j]);
}

// One workgroup owns 16 hidden columns: computes i/f/g/o gate tiles for those
// columns (64x16 each) via bf16 WMMA over K=2048 ([x|h] concat), then applies
// the cell nonlinearity and writes c (f32, in place) and h (bf16) to the
// next-step input buffers.
__global__ void __launch_bounds__(256)
k_lstm(const bf16_t* __restrict__ xh, const bf16_t* __restrict__ wcat,
       const float* __restrict__ bias, float* __restrict__ c,
       bf16_t* __restrict__ hdst0, bf16_t* __restrict__ hdst1) {
  __shared__ float gtile[4][B_DIM][16];
  const int jg   = blockIdx.x;              // 0..63 column group
  const int tid  = threadIdx.x;
  const int lane = tid & 31;
  const int w    = tid >> 5;                // 8 waves
  const int m0   = (w & 3) << 4;            // row tile
  const int g0   = (w >> 2) << 1;           // gate pair {g0, g0+1}
  const int nB0  = g0 * H_DIM + jg * 16;
  const int nB1  = (g0 + 1) * H_DIM + jg * 16;

  v8f acc0 = {}; v8f acc1 = {};
  for (int k0 = 0; k0 < KCAT; k0 += 32) {
    v16bf a  = load_a_frag(xh, KCAT, m0, k0, lane);
    v16bf b0 = load_b_frag(wcat, KCAT, nB0, k0, lane);
    v16bf b1 = load_b_frag(wcat, KCAT, nB1, k0, lane);
    acc0 = wmma_bf16(a, b0, acc0);
    acc1 = wmma_bf16(a, b1, acc1);
  }
  const int hi = lane >> 4, cn = lane & 15;
#pragma unroll
  for (int v = 0; v < 8; ++v) {
    int r = m0 + v + (hi << 3);             // D layout: VGPR v -> M = v (+8 hi)
    gtile[g0][r][cn]     = acc0[v];
    gtile[g0 + 1][r][cn] = acc1[v];
  }
  __syncthreads();

  const int r  = tid >> 2;
  const int n0 = (tid & 3) << 2;
#pragma unroll
  for (int q = 0; q < 4; ++q) {
    int n = n0 + q, colG = jg * 16 + n;
    float iv = gtile[0][r][n] + bias[0 * H_DIM + colG];
    float fv = gtile[1][r][n] + bias[1 * H_DIM + colG];
    float gv = gtile[2][r][n] + bias[2 * H_DIM + colG];
    float ov = gtile[3][r][n] + bias[3 * H_DIM + colG];
    float cP = c[(size_t)r * H_DIM + colG];
    float c2 = sigf(fv) * cP + sigf(iv) * tanhf(gv);
    float h2 = sigf(ov) * tanhf(c2);
    c[(size_t)r * H_DIM + colG] = c2;
    bf16_t hb = f2bf(h2);
    hdst0[(size_t)r * KCAT + colG] = hb;
    if (hdst1) hdst1[(size_t)r * KCAT + colG] = hb;
  }
}

// 128 logit columns per workgroup: bf16 WMMA GEMM + bias, stream logits to
// d_out, and emit per-block argmax partials (first-index tie-break).
__global__ void __launch_bounds__(256)
k_logits(const bf16_t* __restrict__ h1, const bf16_t* __restrict__ wout,
         const float* __restrict__ bout, float* __restrict__ out,
         float* __restrict__ pval, int* __restrict__ pidx) {
  __shared__ float slog[B_DIM][LCOLS];
  __shared__ float sbv[B_DIM][4];
  __shared__ int   sbi[B_DIM][4];
  const int colBase = blockIdx.x * LCOLS;   // 250 blocks
  const int tid = threadIdx.x, lane = tid & 31, w = tid >> 5;
  const int m0 = (w >> 1) << 4;
  const int nq = (w & 1) << 2;              // 4 N-tiles per wave

  v8f acc[4] = {{}, {}, {}, {}};
  for (int k0 = 0; k0 < H_DIM; k0 += 32) {
    v16bf a = load_a_frag(h1, KCAT, m0, k0, lane);
#pragma unroll
    for (int t = 0; t < 4; ++t) {
      v16bf b = load_b_frag(wout, H_DIM, colBase + ((nq + t) << 4), k0, lane);
      acc[t] = wmma_bf16(a, b, acc[t]);
    }
  }
  const int hi = lane >> 4, cn = lane & 15;
#pragma unroll
  for (int t = 0; t < 4; ++t) {
    int cL = ((nq + t) << 4) + cn;
    float bb = bout[colBase + cL];
#pragma unroll
    for (int v = 0; v < 8; ++v) slog[m0 + v + (hi << 3)][cL] = acc[t][v] + bb;
  }
  __syncthreads();

  const int r = tid >> 2, seg = tid & 3;
  float* orow = out + (size_t)r * ((size_t)NSTEPS * V_DIM) + colBase + seg * 32;
  const float* srow = &slog[r][seg * 32];
#pragma unroll
  for (int j = 0; j < 32; j += 4)
    *(float4*)(orow + j) = make_float4(srow[j], srow[j + 1], srow[j + 2], srow[j + 3]);

  float bv = -3.4e38f; int bi = 0x7fffffff;
  for (int j = 0; j < 32; ++j) {
    float v = srow[j]; int gi = colBase + seg * 32 + j;
    if (v > bv || (v == bv && gi < bi)) { bv = v; bi = gi; }
  }
  sbv[r][seg] = bv; sbi[r][seg] = bi;
  __syncthreads();
  if (tid < B_DIM) {
    float v = sbv[tid][0]; int gi = sbi[tid][0];
#pragma unroll
    for (int s = 1; s < 4; ++s) {
      float v2 = sbv[tid][s]; int i2 = sbi[tid][s];
      if (v2 > v || (v2 == v && i2 < gi)) { v = v2; gi = i2; }
    }
    pval[(size_t)tid * NPART + blockIdx.x] = v;
    pidx[(size_t)tid * NPART + blockIdx.x] = gi;
  }
}

__global__ void k_argmax(const float* __restrict__ pval, const int* __restrict__ pidx,
                         int* __restrict__ last) {
  __shared__ float sv[256];
  __shared__ int   si[256];
  const int b = blockIdx.x, tid = threadIdx.x;
  float bv = -3.4e38f; int bi = 0x7fffffff;
  for (int i = tid; i < NPART; i += 256) {
    float v = pval[(size_t)b * NPART + i];
    int  gi = pidx[(size_t)b * NPART + i];
    if (v > bv || (v == bv && gi < bi)) { bv = v; bi = gi; }
  }
  sv[tid] = bv; si[tid] = bi;
  __syncthreads();
  for (int s = 128; s > 0; s >>= 1) {
    if (tid < s) {
      float v2 = sv[tid + s]; int i2 = si[tid + s];
      if (v2 > sv[tid] || (v2 == sv[tid] && i2 < si[tid])) { sv[tid] = v2; si[tid] = i2; }
    }
    __syncthreads();
  }
  if (tid == 0) last[b] = si[0];
}

// -----------------------------------------------------------------------------

extern "C" void kernel_launch(void* const* d_in, const int* in_sizes, int n_in,
                              void* d_out, int out_size, void* d_ws, size_t ws_size,
                              hipStream_t stream) {
  const float* hidden = (const float*)d_in[0];
  const float* emb    = (const float*)d_in[1];
  const float* Wih0   = (const float*)d_in[2];
  const float* Whh0   = (const float*)d_in[3];
  const float* bih0   = (const float*)d_in[4];
  const float* bhh0   = (const float*)d_in[5];
  const float* Wih1   = (const float*)d_in[6];
  const float* Whh1   = (const float*)d_in[7];
  const float* bih1   = (const float*)d_in[8];
  const float* bhh1   = (const float*)d_in[9];
  const float* Wout   = (const float*)d_in[10];
  const float* bout   = (const float*)d_in[11];
  float* out = (float*)d_out;
  (void)in_sizes; (void)n_in; (void)out_size; (void)ws_size;

  char* ws = (char*)d_ws;
  size_t off = 0;
  auto take = [&](size_t bytes) -> char* {
    off = (off + 255) & ~(size_t)255;
    char* p = ws + off; off += bytes; return p;
  };
  bf16_t* wcat0 = (bf16_t*)take((size_t)G4 * KCAT * 2);      // 16 MB
  bf16_t* wcat1 = (bf16_t*)take((size_t)G4 * KCAT * 2);      // 16 MB
  bf16_t* woutb = (bf16_t*)take((size_t)V_DIM * H_DIM * 2);  // 62.5 MB
  float*  b0    = (float*)take((size_t)G4 * 4);
  float*  b1    = (float*)take((size_t)G4 * 4);
  bf16_t* xh0[2] = {(bf16_t*)take((size_t)B_DIM * KCAT * 2),
                    (bf16_t*)take((size_t)B_DIM * KCAT * 2)};
  bf16_t* xh1[2] = {(bf16_t*)take((size_t)B_DIM * KCAT * 2),
                    (bf16_t*)take((size_t)B_DIM * KCAT * 2)};
  float* c0 = (float*)take((size_t)B_DIM * H_DIM * 4);
  float* c1 = (float*)take((size_t)B_DIM * H_DIM * 4);
  int*  last = (int*)take(B_DIM * 4);
  float* pval = (float*)take((size_t)B_DIM * NPART * 4);
  int*   pidx = (int*)take((size_t)B_DIM * NPART * 4);

  // one-time (per launch) bf16 weight prep
  k_conv_wcat<<<G4, 256, 0, stream>>>(Wih0, Whh0, wcat0);
  k_conv_wcat<<<G4, 256, 0, stream>>>(Wih1, Whh1, wcat1);
  k_conv_wout<<<4096, 256, 0, stream>>>(Wout, woutb, V_DIM * H_DIM);
  k_bias<<<16, 256, 0, stream>>>(bih0, bhh0, b0, bih1, bhh1, b1);
  k_init<<<(B_DIM * H_DIM + 255) / 256, 256, 0, stream>>>(hidden, xh0[0], xh1[0],
                                                          c0, c1, last);

  for (int s = 0; s < NSTEPS; ++s) {
    const int p = s & 1, q = p ^ 1;
    // x = emb[last] -> xh0[p][:, 0:H]
    k_gather<<<B_DIM, 256, 0, stream>>>(emb, last, xh0[p]);
    // layer 0: reads xh0[p]; h0 -> xh1[p][:,0:H] (this step) and xh0[q][:,H:2H] (next step)
    k_lstm<<<H_DIM / 16, 256, 0, stream>>>(xh0[p], wcat0, b0, c0,
                                           xh1[p], xh0[q] + H_DIM);
    // layer 1: reads xh1[p]; h1 -> xh1[q][:,H:2H] (next step + logits input)
    k_lstm<<<H_DIM / 16, 256, 0, stream>>>(xh1[p], wcat1, b1, c1,
                                           xh1[q] + H_DIM, nullptr);
    // logits for this step, plus per-block argmax partials
    k_logits<<<NPART, 256, 0, stream>>>(xh1[q] + H_DIM, woutb, bout,
                                        out + (size_t)s * V_DIM, pval, pidx);
    k_argmax<<<B_DIM, 256, 0, stream>>>(pval, pidx, last);
  }
}
